// LatticeStyleConv_73143293051386
// MI455X (gfx1250) — matrix operand order repaired and checked by reading, not compiled, not run on hardware
//
#include <hip/hip_runtime.h>
#include <hip/hip_bf16.h>

typedef __attribute__((ext_vector_type(16))) _Float16 v16h;
typedef __attribute__((ext_vector_type(8)))  _Float16 v8h;
typedef __attribute__((ext_vector_type(8)))  float    v8f;

#define CI 256
#define CO 256
#define BS 8
#define HH 128
#define WW 128
#define NS 5
#define K_TOTAL (NS * CI)        // 1280
#define NCHUNK  (K_TOTAL / 32)   // 40
#define LDSN    (4 * 32 * 16)    // halves per LDS buffer (4 n-subtiles)

// ---------------------------------------------------------------------------
// Kernel 1: StyleGAN2 modulation + demodulation -> f16 weights
//   w[b,o,ci,s] = weights[o,ci,s] * (y[b,ci] + 1)
//   d[b,o]      = rsqrt(sum_{ci,s} w^2 + 1e-6)
//   wf16[(b*CO+o)][s*256+ci] = (f16)(w * d)      (K = s*256+ci ordering)
// ---------------------------------------------------------------------------
__global__ __launch_bounds__(256) void modulate_weights_f16(
    const float* __restrict__ weights,   // [co][ci][5]
    const float* __restrict__ y,         // [bs][ci]
    _Float16* __restrict__ wf16)         // [bs*co][1280]
{
  const int o  = blockIdx.x;
  const int b  = blockIdx.y;
  const int ci = threadIdx.x;            // 256 threads, one per ci

  const float ym = y[b * CI + ci] + 1.0f;
  float w[NS];
  float ss = 0.0f;
#pragma unroll
  for (int s = 0; s < NS; ++s) {
    const float v = weights[(o * CI + ci) * NS + s] * ym;
    w[s] = v;
    ss += v * v;
  }

  __shared__ float red[256];
  red[ci] = ss;
  __syncthreads();
  for (int off = 128; off > 0; off >>= 1) {
    if (ci < off) red[ci] += red[ci + off];
    __syncthreads();
  }
  const float d = rsqrtf(red[0] + 1e-6f);

  _Float16* row = wf16 + (size_t)(b * CO + o) * K_TOTAL;
#pragma unroll
  for (int s = 0; s < NS; ++s)
    row[s * CI + ci] = (_Float16)(w[s] * d);
}

// ---------------------------------------------------------------------------
// Kernel 2: implicit-GEMM stencil conv, v_wmma_f32_16x16x32_f16, software
// pipelined with register-staged LDS double buffering (1 barrier / K-chunk).
//   Block: 512 threads = 16 waves. Block tile: ALL 256 co x 64 pixels, so
//   each B tile (32K x 64N) is staged exactly once per (batch, pixel tile).
//   K = s*256 + ci, chunks of 32 (each chunk = one stencil shift).
// ---------------------------------------------------------------------------
__global__ __launch_bounds__(512) void lattice_conv_wmma(
    const float* __restrict__ x,         // [bs][ci][128][128] f32
    const _Float16* __restrict__ wf16,   // [bs*co][1280] f16
    const int* __restrict__ stencil,     // [5] flat 3x3 indices
    float* __restrict__ out)             // [bs][co][128][128] f32
{
  // double buffer: 2 x (4 n-subtiles x 32 lanes x 16 halves), 8 KB total
  __shared__ __align__(32) _Float16 ldsB[2 * LDSN];

  const int tid  = threadIdx.x;
  const int lane = tid & 31;
  const int wave = tid >> 5;                       // 0..15

  const int b    = blockIdx.y;
  const int m0   = wave * 16;                      // co tile base for wave
  const int p0   = blockIdx.x * 64;                // 64 pixels, single image row
  const int hrow = p0 >> 7;                        // image row
  const int cb   = p0 & 127;                       // column base (0 or 64)

  const float* xb = x + ((size_t)b << 22);         // b * 256 * 128 * 128

  // Pack (dy+1, dx+1) per stencil tap into nibbles of one scalar register.
  unsigned packed = 0;
#pragma unroll
  for (int s = 0; s < NS; ++s) {
    const int si = stencil[s];
    packed |= (unsigned)(((si / 3) << 2) | (si % 3)) << (4 * s);
  }

  // A operand addressing (16-bit A 16x32 layout):
  //   lanes 0-15 row M=lane,    halves = K{0..7, 16..23}
  //   lanes16-31 row M=lane-16, halves = K{8..15, 24..31}
  const _Float16* arow =
      wf16 + (size_t)(b * CO + m0 + (lane & 15)) * K_TOTAL;
  const int sel = (lane < 16) ? 0 : 8;

  const int nstage = tid & 63;                     // staging: pixel within tile
  const int kbase  = tid >> 6;                     // staging: K phase 0..7

  // --- pipeline helpers -----------------------------------------------------
  float vst[4];                                    // register stage (4/thread)

  auto loadChunk = [&](int chunk) {
    const int s   = chunk >> 3;
    const unsigned nib = (packed >> (4 * s)) & 15u;
    const int dys = (int)(nib >> 2) - 1;
    const int dxs = (int)(nib & 3) - 1;
    const int hs  = hrow + dys;
    const int ccol = cb + nstage + dxs;
    const bool ok = (hs >= 0) & (hs < HH) & (ccol >= 0) & (ccol < WW);
    const int cib = (chunk & 7) << 5;              // ci base of chunk
    const size_t base = ((size_t)cib << 14) + (size_t)((hs << 7) + ccol);
#pragma unroll
    for (int i = 0; i < 4; ++i) {
      const int kl = kbase + i * 8;                // 0..31
      vst[i] = ok ? xb[base + ((size_t)kl << 14)] : 0.0f;
    }
  };

  auto storeChunk = [&](_Float16* buf) {
    const int nt = nstage >> 4;                    // pixel sub-tile 0..3
#pragma unroll
    for (int i = 0; i < 4; ++i) {
      const int kl = kbase + i * 8;
      const int ln = (nstage & 15) | ((kl & 8) << 1);    // dest lane in B layout
      buf[((nt * 32 + ln) << 4) | ((kl & 7) | ((kl & 16) >> 1))] = (_Float16)vst[i];
    }
  };

  v8f acc[4] = {};                                 // f32 accumulators

  // --- software-pipelined main loop ------------------------------------------
  loadChunk(0);
  storeChunk(ldsB);
  int bufsel = 0;

  for (int chunk = 0; chunk < NCHUNK; ++chunk) {
    const bool more = (chunk + 1) < NCHUNK;
    if (more) loadChunk(chunk + 1);                // overlaps barrier + wmma

    __syncthreads();                               // buf[bufsel] ready

    const _Float16* buf = ldsB + bufsel * LDSN;
    const int k0 = chunk * 32;
    const v8h a0 = *(const v8h*)(arow + k0 + sel);
    const v8h a1 = *(const v8h*)(arow + k0 + 16 + sel);
    const v16h av = __builtin_shufflevector(
        a0, a1, 0, 1, 2, 3, 4, 5, 6, 7, 8, 9, 10, 11, 12, 13, 14, 15);
#pragma unroll
    for (int nt = 0; nt < 4; ++nt) {
      const v16h bv = *(const v16h*)(buf + ((nt * 32 + lane) << 4));
      acc[nt] = __builtin_amdgcn_wmma_f32_16x16x32_f16(
          /*neg_a=*/false, av, /*neg_b=*/false, bv,
          /*c_mod=*/(short)0, acc[nt], /*reuse_a=*/false, /*reuse_b=*/false);
    }

    if (more) storeChunk(ldsB + (bufsel ^ 1) * LDSN);
    bufsel ^= 1;
  }

  // Epilogue per C layout: VGPR r -> M = r (lanes 0-15) / 8+r (lanes 16-31),
  // N = lane % 16. Non-temporal: output is write-once, keep it out of L2 so
  // x (per-batch 16 MB) stays resident.
  const int nloc = lane & 15;
  const int mhi  = (lane >> 4) << 3;
#pragma unroll
  for (int nt = 0; nt < 4; ++nt)
#pragma unroll
    for (int r = 0; r < 8; ++r)
      __builtin_nontemporal_store(
          acc[nt][r],
          out + ((size_t)(b * CO + m0 + mhi + r) << 14) + p0 + nt * 16 + nloc);
}

// ---------------------------------------------------------------------------
extern "C" void kernel_launch(void* const* d_in, const int* in_sizes, int n_in,
                              void* d_out, int out_size, void* d_ws,
                              size_t ws_size, hipStream_t stream) {
  const float* x       = (const float*)d_in[0];  // [8,256,128,128]
  const float* y       = (const float*)d_in[1];  // [8,256]
  const float* weights = (const float*)d_in[2];  // [256,256,5]
  const int* stencil   = (const int*)d_in[3];    // [5]
  float* out           = (float*)d_out;          // [8,256,128,128]
  _Float16* wf16       = (_Float16*)d_ws;        // 8*256*1280 f16 = 5.25 MB

  (void)in_sizes; (void)n_in; (void)out_size; (void)ws_size;

  // 1) modulated + demodulated f16 weights, K = s*256+ci layout
  modulate_weights_f16<<<dim3(CO, BS), 256, 0, stream>>>(weights, y, wf16);

  // 2) implicit GEMM: grid = (HW/64 pixel tiles, batch); 512-thread blocks
  lattice_conv_wmma<<<dim3((HH * WW) / 64, BS), 512, 0, stream>>>(
      x, wf16, stencil, out);
}
